// MoD_19954418057569
// MI455X (gfx1250) — matrix operand order repaired and checked by reading, not compiled, not run on hardware
//
#include <hip/hip_runtime.h>
#include <hip/hip_bf16.h>
#include <math.h>

#define B_    4
#define S_    4096
#define D_    2048
#define DFF_  8192
#define K_    512
#define BS_   (B_ * S_)
#define BK_   (B_ * K_)

typedef __attribute__((ext_vector_type(16))) __bf16 v16bf;
typedef __attribute__((ext_vector_type(8)))  __bf16 v8bf;
typedef __attribute__((ext_vector_type(8)))  float  v8f;

// ---------- helpers ----------
__device__ __forceinline__ unsigned short f32_to_bf16(float f) {
    unsigned int u = __float_as_uint(f);
    u = (u + 0x7FFFu + ((u >> 16) & 1u)) >> 16;   // RNE
    return (unsigned short)u;
}

// order-preserving float -> uint key (larger float => larger key)
__device__ __forceinline__ unsigned int f2key(float f) {
    unsigned int u = __float_as_uint(f);
    return (u & 0x80000000u) ? ~u : (u | 0x80000000u);
}

__device__ __forceinline__ float gelu_tanh(float x) {
    const float c0 = 0.7978845608028654f;   // sqrt(2/pi)
    const float c1 = 0.044715f;
    float x3 = x * x * x;
    return 0.5f * x * (1.0f + tanhf(c0 * (x + c1 * x3)));
}

// ---------- 1) router: w[b,s] = x[b,s,:] . W_r + b_r  (one wave per row) ----------
__global__ __launch_bounds__(256) void router_kernel(
    const float* __restrict__ x, const float* __restrict__ Wr,
    const float* __restrict__ br, float* __restrict__ wts)
{
    const int row  = blockIdx.x * 8 + (threadIdx.x >> 5);
    const int lane = threadIdx.x & 31;
    const float4* xr = (const float4*)(x + (size_t)row * D_);
    const float4* wr = (const float4*)Wr;
    float s = 0.f;
    for (int i = lane; i < D_ / 4; i += 32) {
        float4 a = xr[i], b = wr[i];
        s += a.x * b.x + a.y * b.y + a.z * b.z + a.w * b.w;
    }
    for (int off = 16; off; off >>= 1) s += __shfl_down(s, off, 32);
    if (lane == 0) wts[row] = s + br[0];
}

// ---------- 2) per-batch K-th largest via 32-step bitwise search on uint keys ----------
__global__ __launch_bounds__(256) void topk_threshold_kernel(
    const float* __restrict__ wts, unsigned int* __restrict__ thr_key)
{
    const int b = blockIdx.x;
    const float* w = wts + (size_t)b * S_;
    __shared__ int red[8];
    __shared__ int totalSh;
    unsigned int T = 0;
    for (int bit = 31; bit >= 0; --bit) {
        const unsigned int cand = T | (1u << bit);
        int cnt = 0;
        for (int i = threadIdx.x; i < S_; i += 256)
            cnt += (f2key(w[i]) >= cand) ? 1 : 0;
        for (int off = 16; off; off >>= 1) cnt += __shfl_down(cnt, off, 32);
        if ((threadIdx.x & 31) == 0) red[threadIdx.x >> 5] = cnt;
        __syncthreads();
        if (threadIdx.x == 0) {
            int t = 0;
            for (int i = 0; i < 8; ++i) t += red[i];
            totalSh = t;
        }
        __syncthreads();
        if (totalSh >= K_) T = cand;   // max T with count(key >= T) >= K
    }
    if (threadIdx.x == 0) thr_key[b] = T;
}

// ---------- 3) init slots ----------
__global__ void init_kernel(int* __restrict__ tok, int* __restrict__ cnt) {
    int i = blockIdx.x * blockDim.x + threadIdx.x;
    if (i < BK_) tok[i] = -1;
    if (i < B_) cnt[i] = 0;
}

// ---------- 4) compact strictly-greater tokens (slot order irrelevant) ----------
__global__ __launch_bounds__(256) void compact_kernel(
    const float* __restrict__ wts, const unsigned int* __restrict__ thr_key,
    int* __restrict__ cnt, int* __restrict__ tok, float* __restrict__ wsel)
{
    const int g = blockIdx.x * blockDim.x + threadIdx.x;   // over B*S
    const int b = g >> 12;                                  // / S_
    const int s = g & (S_ - 1);
    const float w = wts[g];
    if (f2key(w) > thr_key[b]) {                            // strictly greater => <= K-1 per batch
        int slot = atomicAdd(&cnt[b], 1);
        tok[b * K_ + slot] = s;
        wsel[b * K_ + slot] = w;
    }
}

// ---------- 5) gather selected tokens, fp32 -> bf16 (padding rows = 0) ----------
__global__ __launch_bounds__(256) void gather_kernel(
    const float* __restrict__ x, const int* __restrict__ tok,
    unsigned short* __restrict__ Xg)
{
    const int row = blockIdx.x;           // 0 .. BK_-1
    const int d0  = threadIdx.x * 8;      // 256 * 8 = 2048
    const int tv  = tok[row];
    uint4 pk;
    if (tv >= 0) {
        const int b = row >> 9;           // / K_
        const float4* src = (const float4*)(x + ((size_t)b * S_ + tv) * D_ + d0);
        float4 a = src[0], c = src[1];
        pk.x = (unsigned)f32_to_bf16(a.x) | ((unsigned)f32_to_bf16(a.y) << 16);
        pk.y = (unsigned)f32_to_bf16(a.z) | ((unsigned)f32_to_bf16(a.w) << 16);
        pk.z = (unsigned)f32_to_bf16(c.x) | ((unsigned)f32_to_bf16(c.y) << 16);
        pk.w = (unsigned)f32_to_bf16(c.z) | ((unsigned)f32_to_bf16(c.w) << 16);
    } else {
        pk = make_uint4(0u, 0u, 0u, 0u);
    }
    *(uint4*)(Xg + (size_t)row * D_ + d0) = pk;
}

// ---------- 6) LDS-tiled transpose + fp32->bf16: out[C][R] = bf16(in[R][C]) ----------
__global__ __launch_bounds__(256) void transpose_bf16_kernel(
    const float* __restrict__ in, unsigned short* __restrict__ out, int R, int C)
{
    __shared__ float tile[32][33];
    const int c0 = blockIdx.x * 32, r0 = blockIdx.y * 32;
    const int tx = threadIdx.x & 31, ty = threadIdx.x >> 5;   // ty: 0..7
    #pragma unroll
    for (int i = 0; i < 32; i += 8)
        tile[ty + i][tx] = in[(size_t)(r0 + ty + i) * C + c0 + tx];
    __syncthreads();
    #pragma unroll
    for (int i = 0; i < 32; i += 8)
        out[(size_t)(c0 + ty + i) * R + r0 + tx] = f32_to_bf16(tile[tx][ty + i]);
}

// ---------- 7) out = x (selected rows overwritten later) ----------
__global__ __launch_bounds__(256) void copy_kernel(
    const float* __restrict__ x, float* __restrict__ out)
{
    size_t i = (size_t)blockIdx.x * blockDim.x + threadIdx.x;
    ((float4*)out)[i] = ((const float4*)x)[i];
}

// ---------- WMMA fragment loaders (bf16, K-contiguous layouts) ----------
// A (16xK, row-major): lane<16 -> row=lane, elems 0..15 = K[k0..k0+15]
//                      lane>=16 -> row=lane-16, K shifted by +16.
__device__ __forceinline__ v16bf load_frag_a(const __bf16* arow, int k) {
    v8bf lo = *reinterpret_cast<const v8bf*>(arow + k);
    v8bf hi = *reinterpret_cast<const v8bf*>(arow + k + 8);
    return __builtin_shufflevector(lo, hi, 0,1,2,3,4,5,6,7,8,9,10,11,12,13,14,15);
}
// B stored transposed Bt[N][K]: lane holds column n=lane&15, K-half by lane>>4 -> one 32B load.
__device__ __forceinline__ v16bf load_frag_b(const __bf16* bcol, int k) {
    return *reinterpret_cast<const v16bf*>(bcol + k);
}

#define WMMA_BF16(a, b, c) \
    __builtin_amdgcn_wmma_f32_16x16x32_bf16(false, (a), false, (b), (short)0, (c), false, false)

// ---------- 8) GEMM1: H = gelu(Xg @ W1 + b1), bf16 out.  M=BK_, N=DFF_, K=D_ ----------
// wave tile 32x64 (2 A frags x 4 B frags -> 8 WMMAs / k-step, 21.3 FLOP/B from L2)
// block = 8 waves as 2(M) x 4(N) -> 64x256 block tile
__global__ __launch_bounds__(256) void gemm1_kernel(
    const unsigned short* __restrict__ Xg,   // [BK_][D_] bf16
    const unsigned short* __restrict__ W1t,  // [DFF_][D_] bf16 (transposed)
    const float* __restrict__ b1,
    unsigned short* __restrict__ H)          // [BK_][DFF_] bf16
{
    const int lane = threadIdx.x & 31;
    const int wave = threadIdx.x >> 5;
    const int m0 = (blockIdx.y << 6) + ((wave >> 2) << 5);   // 32-row wave tile
    const int n0 = (blockIdx.x << 8) + ((wave & 3) << 6);    // 64-col wave tile
    const int half = lane >> 4, l15 = lane & 15;

    const __bf16* A  = reinterpret_cast<const __bf16*>(Xg);
    const __bf16* Bt = reinterpret_cast<const __bf16*>(W1t);
    const __bf16* ar0 = A + (size_t)(m0 + l15) * D_ + half * 16;
    const __bf16* ar1 = A + (size_t)(m0 + 16 + l15) * D_ + half * 16;
    const __bf16* bp0 = Bt + (size_t)(n0 + l15 +  0) * D_ + half * 16;
    const __bf16* bp1 = Bt + (size_t)(n0 + l15 + 16) * D_ + half * 16;
    const __bf16* bp2 = Bt + (size_t)(n0 + l15 + 32) * D_ + half * 16;
    const __bf16* bp3 = Bt + (size_t)(n0 + l15 + 48) * D_ + half * 16;

    v8f acc[2][4] = {};
    for (int k = 0; k < D_; k += 32) {
        if ((k & 127) == 0 && k + 1024 < D_) {      // stream-ahead hints for B (2KB/lane ahead)
            __builtin_prefetch(bp0 + k + 1024, 0, 1);
            __builtin_prefetch(bp1 + k + 1024, 0, 1);
            __builtin_prefetch(bp2 + k + 1024, 0, 1);
            __builtin_prefetch(bp3 + k + 1024, 0, 1);
        }
        v16bf a0 = load_frag_a(ar0, k);
        v16bf a1 = load_frag_a(ar1, k);
        v16bf b0 = load_frag_b(bp0, k);
        v16bf b1v = load_frag_b(bp1, k);
        v16bf b2v = load_frag_b(bp2, k);
        v16bf b3v = load_frag_b(bp3, k);
        acc[0][0] = WMMA_BF16(a0, b0,  acc[0][0]);
        acc[0][1] = WMMA_BF16(a0, b1v, acc[0][1]);
        acc[0][2] = WMMA_BF16(a0, b2v, acc[0][2]);
        acc[0][3] = WMMA_BF16(a0, b3v, acc[0][3]);
        acc[1][0] = WMMA_BF16(a1, b0,  acc[1][0]);
        acc[1][1] = WMMA_BF16(a1, b1v, acc[1][1]);
        acc[1][2] = WMMA_BF16(a1, b2v, acc[1][2]);
        acc[1][3] = WMMA_BF16(a1, b3v, acc[1][3]);
    }
    #pragma unroll
    for (int mi = 0; mi < 2; ++mi) {
        const int rbase = m0 + mi * 16 + half * 8;   // lanes 16-31 hold rows M=8..15
        #pragma unroll
        for (int t = 0; t < 4; ++t) {
            const int col = n0 + t * 16 + l15;
            const float bias = b1[col];
            #pragma unroll
            for (int v = 0; v < 8; ++v) {
                float y = gelu_tanh(acc[mi][t][v] + bias);
                H[(size_t)(rbase + v) * DFF_ + col] = f32_to_bf16(y);
            }
        }
    }
}

// ---------- 9) GEMM2: out[b,tok] = (H @ W2 + b2) * w.  M=BK_, N=D_, K=DFF_ ----------
__global__ __launch_bounds__(256) void gemm2_kernel(
    const unsigned short* __restrict__ H,    // [BK_][DFF_] bf16
    const unsigned short* __restrict__ W2t,  // [D_][DFF_] bf16 (transposed)
    const float* __restrict__ b2,
    const int* __restrict__ tok, const float* __restrict__ wsel,
    float* __restrict__ out)
{
    const int lane = threadIdx.x & 31;
    const int wave = threadIdx.x >> 5;
    const int m0 = (blockIdx.y << 6) + ((wave >> 2) << 5);
    const int n0 = (blockIdx.x << 8) + ((wave & 3) << 6);
    const int half = lane >> 4, l15 = lane & 15;

    const __bf16* A  = reinterpret_cast<const __bf16*>(H);
    const __bf16* Bt = reinterpret_cast<const __bf16*>(W2t);
    const __bf16* ar0 = A + (size_t)(m0 + l15) * DFF_ + half * 16;
    const __bf16* ar1 = A + (size_t)(m0 + 16 + l15) * DFF_ + half * 16;
    const __bf16* bp0 = Bt + (size_t)(n0 + l15 +  0) * DFF_ + half * 16;
    const __bf16* bp1 = Bt + (size_t)(n0 + l15 + 16) * DFF_ + half * 16;
    const __bf16* bp2 = Bt + (size_t)(n0 + l15 + 32) * DFF_ + half * 16;
    const __bf16* bp3 = Bt + (size_t)(n0 + l15 + 48) * DFF_ + half * 16;

    v8f acc[2][4] = {};
    for (int k = 0; k < DFF_; k += 32) {
        if ((k & 127) == 0 && k + 1024 < DFF_) {
            __builtin_prefetch(bp0 + k + 1024, 0, 1);
            __builtin_prefetch(bp1 + k + 1024, 0, 1);
            __builtin_prefetch(bp2 + k + 1024, 0, 1);
            __builtin_prefetch(bp3 + k + 1024, 0, 1);
        }
        v16bf a0 = load_frag_a(ar0, k);
        v16bf a1 = load_frag_a(ar1, k);
        v16bf b0 = load_frag_b(bp0, k);
        v16bf b1v = load_frag_b(bp1, k);
        v16bf b2v = load_frag_b(bp2, k);
        v16bf b3v = load_frag_b(bp3, k);
        acc[0][0] = WMMA_BF16(a0, b0,  acc[0][0]);
        acc[0][1] = WMMA_BF16(a0, b1v, acc[0][1]);
        acc[0][2] = WMMA_BF16(a0, b2v, acc[0][2]);
        acc[0][3] = WMMA_BF16(a0, b3v, acc[0][3]);
        acc[1][0] = WMMA_BF16(a1, b0,  acc[1][0]);
        acc[1][1] = WMMA_BF16(a1, b1v, acc[1][1]);
        acc[1][2] = WMMA_BF16(a1, b2v, acc[1][2]);
        acc[1][3] = WMMA_BF16(a1, b3v, acc[1][3]);
    }
    #pragma unroll
    for (int mi = 0; mi < 2; ++mi) {
        const int rbase = m0 + mi * 16 + half * 8;
        int   tv[8];
        float wv[8];
        #pragma unroll
        for (int v = 0; v < 8; ++v) { tv[v] = tok[rbase + v]; wv[v] = wsel[rbase + v]; }
        #pragma unroll
        for (int t = 0; t < 4; ++t) {
            const int col = n0 + t * 16 + l15;
            const float bias = b2[col];
            #pragma unroll
            for (int v = 0; v < 8; ++v) {
                if (tv[v] < 0) continue;                 // padding row
                const int row = rbase + v;
                const int batch = row >> 9;              // / K_
                out[((size_t)batch * S_ + tv[v]) * D_ + col] = (acc[mi][t][v] + bias) * wv[v];
            }
        }
    }
}

// ---------- launch ----------
extern "C" void kernel_launch(void* const* d_in, const int* in_sizes, int n_in,
                              void* d_out, int out_size, void* d_ws, size_t ws_size,
                              hipStream_t stream) {
    const float* x   = (const float*)d_in[0];
    const float* W_r = (const float*)d_in[1];
    const float* b_r = (const float*)d_in[2];
    const float* W1  = (const float*)d_in[3];
    const float* b1  = (const float*)d_in[4];
    const float* W2  = (const float*)d_in[5];
    const float* b2  = (const float*)d_in[6];
    float* out = (float*)d_out;

    char* p = (char*)d_ws;
    auto alloc = [&](size_t bytes) -> char* {
        char* r = p; p += (bytes + 255) & ~(size_t)255; return r;
    };
    float*          wts  = (float*)alloc((size_t)BS_ * sizeof(float));
    unsigned int*   thrk = (unsigned int*)alloc(B_ * sizeof(unsigned int));
    int*            cnt  = (int*)alloc(B_ * sizeof(int));
    int*            tok  = (int*)alloc((size_t)BK_ * sizeof(int));
    float*          wsel = (float*)alloc((size_t)BK_ * sizeof(float));
    unsigned short* Xg   = (unsigned short*)alloc((size_t)BK_ * D_ * 2);
    unsigned short* W1t  = (unsigned short*)alloc((size_t)DFF_ * D_ * 2);
    unsigned short* W2t  = (unsigned short*)alloc((size_t)D_ * DFF_ * 2);
    unsigned short* H    = (unsigned short*)alloc((size_t)BK_ * DFF_ * 2);

    router_kernel<<<BS_ / 8, 256, 0, stream>>>(x, W_r, b_r, wts);
    topk_threshold_kernel<<<B_, 256, 0, stream>>>(wts, thrk);
    init_kernel<<<(BK_ + 255) / 256, 256, 0, stream>>>(tok, cnt);
    compact_kernel<<<BS_ / 256, 256, 0, stream>>>(wts, thrk, cnt, tok, wsel);
    gather_kernel<<<BK_, 256, 0, stream>>>(x, tok, Xg);
    transpose_bf16_kernel<<<dim3(DFF_ / 32, D_ / 32), 256, 0, stream>>>(W1, W1t, D_, DFF_);
    transpose_bf16_kernel<<<dim3(D_ / 32, DFF_ / 32), 256, 0, stream>>>(W2, W2t, DFF_, D_);
    copy_kernel<<<(unsigned)(((size_t)BS_ * D_ / 4) / 256), 256, 0, stream>>>(x, out);
    gemm1_kernel<<<dim3(DFF_ / 256, BK_ / 64), 256, 0, stream>>>(Xg, W1t, b1, H);
    gemm2_kernel<<<dim3(D_ / 256, BK_ / 64), 256, 0, stream>>>(H, W2t, b2, tok, wsel, out);
}
